// LBPConv_34024730919491
// MI455X (gfx1250) — 8-bit matrix operand order repaired and checked by reading, compiled, not run on hardware
//
#include <hip/hip_runtime.h>
#include <hip/hip_bf16.h>

typedef __attribute__((ext_vector_type(16))) int      v16i;
typedef __attribute__((ext_vector_type(8)))  float    v8f;
typedef __attribute__((ext_vector_type(2)))  unsigned vu2;

#define HH   112
#define WW   112
#define NIMG 16
#define CIN  64
#define COUT 128
#define KDIM 512
#define HWPX (HH * WW)
#define MTOT (NIMG * HWPX)

__device__ __forceinline__ unsigned fp8sign(float v) {
    return v > 0.f ? 0x38u : (v < 0.f ? 0xB8u : 0x00u);
}

__global__ void lbp_prep_A(const float* __restrict__ x, unsigned char* __restrict__ Abits) {
    unsigned t = blockIdx.x * blockDim.x + threadIdx.x;
    int w = t % WW;
    unsigned r = t / WW;
    int h = r % HH;
    r /= HH;
    int c = r % CIN;
    int n = r / CIN;

    const float* xc = x + (size_t)(((n * CIN + c) * HH)) * WW;
    float ctr  = xc[h * WW + w];
    float bias = 1e-8f * ((h & 1) ? 1.f : -1.f) * ((w & 1) ? 1.f : -1.f);

    const int dy[8] = {-1, -1, -1,  0, 0,  1, 1, 1};
    const int dx[8] = {-1,  0,  1, -1, 1, -1, 0, 1};
    unsigned bits = 0;
#pragma unroll
    for (int f = 0; f < 8; ++f) {
        int hh = h + dy[f], ww2 = w + dx[f];
        float nb = (hh >= 0 && hh < HH && ww2 >= 0 && ww2 < WW) ? xc[hh * WW + ww2] : 0.f;
        bits |= ((nb - ctr + bias) < 0.f ? 1u : 0u) << f;
    }

    unsigned p     = (unsigned)n * HWPX + (unsigned)h * WW + (unsigned)w;
    unsigned tile  = p >> 4, mm = p & 15;
    unsigned chunk = (unsigned)c >> 4;
    unsigned c7    = (unsigned)c & 15;
    unsigned lane  = mm + 16u * (c7 & 1);
    unsigned j     = c7 >> 1;
    __builtin_nontemporal_store(
        (unsigned char)bits,
        Abits + (((tile * 4u + chunk) * 32u + lane) << 3) + j);
}

__global__ void lbp_prep_B(const float* __restrict__ wgt, unsigned* __restrict__ Bsw) {
    unsigned t = blockIdx.x * blockDim.x + threadIdx.x;
    unsigned o = t >> 7;
    unsigned g = t & 127;
    unsigned k = g * 4u;

    const float* wp = wgt + (size_t)o * KDIM + k;
    unsigned d = 0;
#pragma unroll
    for (int j = 0; j < 4; ++j) d |= fp8sign(wp[j]) << (8 * j);

    unsigned ntile = o >> 4, nn = o & 15;
    unsigned chunk = k >> 7;
    unsigned k7    = k & 127;
    unsigned blk   = k7 >> 5;
    unsigned k5    = k7 & 31;
    unsigned lane  = nn + 16u * ((k7 >> 3) & 1);
    unsigned vgpr  = 2u * (k7 >> 4) + ((k7 & 7) >> 2);
    Bsw[((ntile * 4u + chunk) * 32u + lane) * 16u + vgpr] = d;
}

__global__ void __launch_bounds__(256) lbp_gemm(const unsigned char* __restrict__ Abits,
                                                const uint4* __restrict__ Bsw4,
                                                float* __restrict__ out) {
    __shared__ uint4 ldsB[4096];

#pragma unroll
    for (int j = 0; j < 16; ++j)
        ldsB[j * 256 + threadIdx.x] = Bsw4[j * 256 + threadIdx.x];
    __syncthreads();

    const unsigned lane = threadIdx.x & 31;
    const unsigned wv   = threadIdx.x >> 5;
    const unsigned tile = blockIdx.x * 8u + wv;

    v8f acc[8] = {};
    const unsigned* ldsW = (const unsigned*)ldsB;

#pragma unroll
    for (int kk = 0; kk < 4; ++kk) {
        vu2 pk = __builtin_nontemporal_load(
            (const vu2*)(Abits + (((size_t)(tile * 4u + kk) * 32u + lane) << 3)));
        v16i a;
#pragma unroll
        for (int j = 0; j < 8; ++j) {
            unsigned byt = ((j < 4 ? pk.x : pk.y) >> (8 * (j & 3))) & 0xFFu;
            unsigned lo = 0x38383838u | ((((byt & 0xFu) * 0x00204081u) & 0x01010101u) << 7);
            unsigned hi = 0x38383838u | ((((byt >> 4)   * 0x00204081u) & 0x01010101u) << 7);
            a[2 * j]     = (int)lo;
            a[2 * j + 1] = (int)hi;
        }
#pragma unroll
        for (int nt = 0; nt < 8; ++nt) {
            v16i b = *(const v16i*)(ldsW + ((((unsigned)nt * 4u + kk) * 32u + lane) << 4));
            acc[nt] = __builtin_amdgcn_wmma_f32_16x16x128_fp8_fp8(
                a, b, (short)0, acc[nt], false, false);
        }
    }

    __syncthreads();

    float* tbuf = ((float*)ldsB) + wv * 256;
    unsigned p0   = tile * 16u;
    unsigned nimg = p0 / HWPX;
    unsigned hw   = p0 % HWPX;
    float* obase  = out + (size_t)nimg * COUT * HWPX + hw + (lane & 15);

#pragma unroll
    for (int nt = 0; nt < 8; ++nt) {
#pragma unroll
        for (int r2 = 0; r2 < 8; ++r2)
            tbuf[(lane & 15) * 16 + r2 + 8 * (lane >> 4)] = acc[nt][r2];
        __builtin_amdgcn_wave_barrier();
        asm volatile("" ::: "memory");
#pragma unroll
        for (int j = 0; j < 8; ++j) {
            unsigned ol = 2u * j + (lane >> 4);
            __builtin_nontemporal_store(tbuf[ol * 16 + (lane & 15)],
                                        obase + (size_t)(nt * 16 + ol) * HWPX);
        }
        __builtin_amdgcn_wave_barrier();
        asm volatile("" ::: "memory");
    }
}

extern "C" void kernel_launch(void* const* d_in, const int* in_sizes, int n_in,
                              void* d_out, int out_size, void* d_ws, size_t ws_size,
                              hipStream_t stream) {
    const float* x   = (const float*)d_in[0];
    const float* wgt = (const float*)d_in[1];
    float* out = (float*)d_out;

    unsigned char* ws    = (unsigned char*)d_ws;
    unsigned*      Bsw   = (unsigned*)ws;
    unsigned char* Abits = ws + 65536;

    lbp_prep_B<<<64, 256, 0, stream>>>(wgt, Bsw);
    lbp_prep_A<<<(MTOT * CIN) / 256, 256, 0, stream>>>(x, Abits);
    lbp_gemm<<<(MTOT / 16) / 8, 256, 0, stream>>>(
        Abits, (const uint4*)Bsw, out);
}
